// ModulatedConv2d_73658689126706
// MI455X (gfx1250) — matrix operand order repaired and checked by reading, not compiled, not run on hardware
//
#include <hip/hip_runtime.h>
#include <stdint.h>
#include <stddef.h>

#define B_    16
#define CIN   512
#define COUT  512
#define HH    64
#define WW    64
#define SDIM  512

typedef __attribute__((ext_vector_type(16))) __bf16 v16bf;
typedef __attribute__((ext_vector_type(8)))  float  v8f;
typedef __attribute__((ext_vector_type(4)))  unsigned int v4u;
typedef __attribute__((ext_vector_type(8)))  int v8i;
typedef __attribute__((ext_vector_type(4)))  int v4i;

struct U4x2 { uint4 lo, hi; };
union FragCast { U4x2 u; v16bf v; };

__device__ inline unsigned short f2bf(float f) {
  union { float f; uint32_t u; } v; v.f = f;
  uint32_t r = v.u + 0x7FFFu + ((v.u >> 16) & 1u);   // round-to-nearest-even
  return (unsigned short)(r >> 16);
}

// s[b][i] = sum_j style[b][j] * style_w[i][j] * (1/sqrt(SDIM)) + style_b[i]
__global__ void k_style(const float* __restrict__ style, const float* __restrict__ style_w,
                        const float* __restrict__ style_b, float* __restrict__ s) {
  int t = blockIdx.x * blockDim.x + threadIdx.x;
  if (t >= B_ * CIN) return;
  int b = t / CIN, i = t % CIN;
  const float lin = 0.04419417382415922f; // 1/sqrt(512)
  const float* st = style + b * SDIM;
  const float* w  = style_w + i * SDIM;
  float acc = 0.f;
  for (int j = 0; j < SDIM; ++j) acc += st[j] * w[j];
  s[t] = acc * lin + style_b[i];
}

// wsq[o][i] = sum over 9 taps of weight^2
__global__ void k_wsq(const float* __restrict__ weight, float* __restrict__ wsq) {
  int t = blockIdx.x * blockDim.x + threadIdx.x;
  if (t >= COUT * CIN) return;
  const float* w = weight + (size_t)t * 9;
  float a = 0.f;
  for (int k = 0; k < 9; ++k) a += w[k] * w[k];
  wsq[t] = a;
}

// dcoef[b][o] = rsqrt(cs^2 * sum_i s[b,i]^2 * wsq[o,i] + 1e-8)
__global__ void k_dcoef(const float* __restrict__ s, const float* __restrict__ wsq,
                        float* __restrict__ dcoef) {
  int t = blockIdx.x * blockDim.x + threadIdx.x;
  if (t >= B_ * COUT) return;
  int b = t / COUT, o = t % COUT;
  const float* sb = s + b * CIN;
  const float* wo = wsq + o * CIN;
  float a = 0.f;
  for (int i = 0; i < CIN; ++i) { float si = sb[i]; a += si * si * wo[i]; }
  const float cs2 = 1.0f / (512.f * 9.f); // conv_scale^2
  dcoef[t] = rsqrtf(a * cs2 + 1e-8f);
}

// xbT[b][y][x][i] = bf16(x[b][i][y][x])  (cin innermost => contiguous K fragments)
__global__ void k_xT(const float* __restrict__ x, unsigned short* __restrict__ xbT) {
  size_t t = (size_t)blockIdx.x * blockDim.x + threadIdx.x;
  if (t >= (size_t)B_ * CIN * HH * WW) return;
  int i  = (int)(t % CIN);
  size_t r = t / CIN;
  int xc = (int)(r % WW); r /= WW;
  int y  = (int)(r % HH);
  int b  = (int)(r / HH);
  float v = x[(((size_t)(b * CIN + i)) * HH + y) * WW + xc];
  xbT[t] = f2bf(v);
}

// wm: modulated+demodulated bf16 weights in WMMA A-fragment order:
// [b][mt(8)][kcc(16)][tap(9)][mf(4)][lane(32)][e(16)]  (contiguous k = 16*(lane/16) + e, same as B fragment)
__global__ void k_modw(const float* __restrict__ weight, const float* __restrict__ s,
                       const float* __restrict__ dcoef, uint4* __restrict__ wm) {
  int t = blockIdx.x * blockDim.x + threadIdx.x;
  const int NT = B_ * 8 * 16 * 9 * 4 * 32;
  if (t >= NT) return;
  int lane = t & 31; int r = t >> 5;
  int mf  = r & 3;  r >>= 2;
  int tap = r % 9;  r /= 9;
  int kcc = r & 15; r >>= 4;
  int mt  = r & 7;
  int b   = r >> 3;
  int o = mt * 64 + mf * 16 + (lane & 15);           // A-matrix: M = lane%16
  const float cs = 0.014731391274719739f;            // 1/sqrt(512*9)
  float dc = dcoef[b * COUT + o] * cs;
  unsigned short buf[16];
  const float* sb = s + b * CIN;
  for (int e = 0; e < 16; ++e) {
    // A-matrix K mapping: contiguous k = 16*(lane/16) + e (matches B fragment)
    int kc = ((lane < 16) ? 0 : 16) + e;
    int cin = kcc * 32 + kc;
    float wv = weight[((size_t)o * CIN + cin) * 9 + tap];
    buf[e] = f2bf(wv * sb[cin] * dc);
  }
  uint4* dst = wm + (size_t)t * 2;
  dst[0] = *(const uint4*)&buf[0];
  dst[1] = *(const uint4*)&buf[8];
}

// Issue one TDM descriptor: contiguous 36864-byte block Global -> LDS.
// D# per CDNA5 ISA: 1-D tile, data_size=8B, tile_dim0=4608.
// 6-arg builtin form (amdgpu-toolchain / clang-23).
__device__ inline void tdm_load_a(const void* gptr, uint32_t lds_byte_addr) {
  uint64_t ga = (uint64_t)(uintptr_t)gptr;
  v4u g0;
  g0.x = 1u;                                            // count=1, user descriptor
  g0.y = lds_byte_addr;                                 // lds_addr [63:32]
  g0.z = (uint32_t)ga;                                  // global_addr[31:0]
  g0.w = (uint32_t)((ga >> 32) & 0x01FFFFFFu) | (2u << 30); // addr[56:32] | type=2
  v8i g1;
  g1[0] = (int)(3u << 16);                              // workgroup_mask=0, data_size=3 (8B)
  g1[1] = (int)(4608u << 16);                           // tensor_dim0[15:0] in bits 63:48
  g1[2] = (int)(1u << 16);                              // tensor_dim0[31:16]=0, tensor_dim1=1
  g1[3] = (int)(4608u << 16);                           // tensor_dim1[31:16]=0, tile_dim0=4608
  g1[4] = 1;                                            // tile_dim1=1, tile_dim2=0
  g1[5] = 4608;                                         // tensor_dim0_stride[31:0]
  g1[6] = 0;                                            // stride0[47:32]=0, stride1[15:0]=0
  g1[7] = 0;
  v4i z4 = {0, 0, 0, 0};
  v8i z8 = {0, 0, 0, 0, 0, 0, 0, 0};
  __builtin_amdgcn_tensor_load_to_lds(g0, g1, z4, z4, z8, 0);
}

// Implicit-GEMM conv: block = (row y, cout-tile of 64, batch b); M=64, N=64, K=512*9.
// 8 waves: wave_m(0..3) x wave_n(0..1); per wave 1 A-frag, 2 B-frags, 2 f32 accum frags.
// A chunks staged via TDM (tensor_load_to_lds), double-buffered, overlapped with WMMA.
__global__ __launch_bounds__(256) void k_conv(const uint4* __restrict__ wm,
                                              const unsigned short* __restrict__ xbT,
                                              float* __restrict__ out) {
  __shared__ uint4 AsmU4[2][2304];                                   // 2 x 36864 B (double buffer)
  __shared__ __attribute__((aligned(16))) unsigned short Bsm[3 * 66 * 32]; // 12672 B: [ky][col+1][cin32]

  const int y   = blockIdx.x;
  const int mt  = blockIdx.y;
  const int b   = blockIdx.z;
  const int tid = threadIdx.x;
  const int lane   = tid & 31;
  const int wid    = tid >> 5;
  const int wave_m = wid >> 1;      // 0..3
  const int wave_n = wid & 1;       // 0..1
  const int halfsel = (lane < 16) ? 0 : 1;
  const int ln16    = lane & 15;

  v8f acc0 = {0.f,0.f,0.f,0.f,0.f,0.f,0.f,0.f};
  v8f acc1 = {0.f,0.f,0.f,0.f,0.f,0.f,0.f,0.f};

  const uint4* wmBase = wm + (size_t)((b * 8 + mt) * 16) * 2304;

  // Prologue: kick off TDM for chunk 0 into buffer 0 (one wave issues; EXEC-independent DMA).
  if (wid == 0) {
    tdm_load_a(wmBase, (uint32_t)(uintptr_t)(void*)&AsmU4[0][0]);
  }

  for (int kcc = 0; kcc < 16; ++kcc) {
    // ---- stage B (overlaps in-flight TDM): rows y-1..y+1, 32 cins, zero-padded cols ----
    for (int t = tid; t < 198; t += 256) {
      if (t < 192) {
        int ry = t / 64, c = t % 64;
        int gy = y + ry - 1;
        uint4* dst = (uint4*)&Bsm[(ry * 66 + (c + 1)) * 32];
        if ((unsigned)gy < 64u) {
          const uint4* sp = (const uint4*)(xbT + ((size_t)((b * 64 + gy) * 64 + c)) * 512 + kcc * 32);
          dst[0] = sp[0]; dst[1] = sp[1]; dst[2] = sp[2]; dst[3] = sp[3];
        } else {
          uint4 zz = {0u,0u,0u,0u};
          dst[0] = zz; dst[1] = zz; dst[2] = zz; dst[3] = zz;
        }
      } else {
        int e = t - 192; int ry = e >> 1; int side = e & 1;       // halo columns = 0
        uint4* dst = (uint4*)&Bsm[(ry * 66 + (side ? 65 : 0)) * 32];
        uint4 zz = {0u,0u,0u,0u};
        dst[0] = zz; dst[1] = zz; dst[2] = zz; dst[3] = zz;
      }
    }
    if (wid == 0) __builtin_amdgcn_s_wait_tensorcnt(0);   // A(kcc) DMA complete
    __syncthreads();                                       // A(kcc) + B(kcc) visible to all waves

    // Issue TDM for next chunk into the other buffer; overlaps with this chunk's WMMAs.
    if (wid == 0 && kcc + 1 < 16) {
      tdm_load_a(wmBase + (size_t)(kcc + 1) * 2304,
                 (uint32_t)(uintptr_t)(void*)&AsmU4[(kcc + 1) & 1][0]);
    }

    const uint4* Abuf = &AsmU4[kcc & 1][0];
    for (int tap = 0; tap < 9; ++tap) {
      const int ky = tap / 3, kx = tap % 3;
      FragCast af;
      const uint4* ap = &Abuf[((tap * 4 + wave_m) * 32 + lane) * 2];
      af.u.lo = ap[0]; af.u.hi = ap[1];
      {
        int col = wave_n * 32 + 0 * 16 + ln16 + kx;               // padded index = n + kx
        FragCast bf_;
        const uint4* bp = (const uint4*)&Bsm[(ky * 66 + col) * 32 + halfsel * 16];
        bf_.u.lo = bp[0]; bf_.u.hi = bp[1];
        acc0 = __builtin_amdgcn_wmma_f32_16x16x32_bf16(false, af.v, false, bf_.v,
                                                       (short)0, acc0, false, false);
      }
      {
        int col = wave_n * 32 + 1 * 16 + ln16 + kx;
        FragCast bf_;
        const uint4* bp = (const uint4*)&Bsm[(ky * 66 + col) * 32 + halfsel * 16];
        bf_.u.lo = bp[0]; bf_.u.hi = bp[1];
        acc1 = __builtin_amdgcn_wmma_f32_16x16x32_bf16(false, af.v, false, bf_.v,
                                                       (short)0, acc1, false, false);
      }
    }
    __syncthreads();   // all waves done reading Bsm before next overwrite
  }

  // C/D layout (ISA p.108): N = lane%16, M = r + (lane<16 ? 0 : 8)
  const int obase = mt * 64 + wave_m * 16 + halfsel * 8;
  for (int r = 0; r < 8; ++r) {
    int o = obase + r;
    size_t rowoff = (((size_t)(b * COUT + o)) * HH + y) * WW;
    out[rowoff + wave_n * 32 + 0 * 16 + ln16] = acc0[r];
    out[rowoff + wave_n * 32 + 1 * 16 + ln16] = acc1[r];
  }
}

extern "C" void kernel_launch(void* const* d_in, const int* in_sizes, int n_in,
                              void* d_out, int out_size, void* d_ws, size_t ws_size,
                              hipStream_t stream) {
  const float* x       = (const float*)d_in[0];
  const float* style   = (const float*)d_in[1];
  const float* weight  = (const float*)d_in[2];
  const float* style_w = (const float*)d_in[3];
  const float* style_b = (const float*)d_in[4];
  float* out = (float*)d_out;

  char* ws = (char*)d_ws;
  float* s     = (float*)(ws);                                  //  32 KB
  float* wsq   = (float*)(ws + (64 << 10));                     //   1 MB
  float* dcoef = (float*)(ws + (64 << 10) + (1 << 20));         //  32 KB
  unsigned short* xbT = (unsigned short*)(ws + (2u << 20));     //  64 MiB
  uint4* wm = (uint4*)(ws + (2u << 20) + (64u << 20));          //  72 MiB

  k_style<<<(B_ * CIN + 255) / 256, 256, 0, stream>>>(style, style_w, style_b, s);
  k_wsq<<<(COUT * CIN + 255) / 256, 256, 0, stream>>>(weight, wsq);
  k_dcoef<<<(B_ * COUT + 255) / 256, 256, 0, stream>>>(s, wsq, dcoef);
  {
    size_t n = (size_t)B_ * CIN * HH * WW;
    k_xT<<<(unsigned)((n + 255) / 256), 256, 0, stream>>>(x, xbT);
  }
  {
    int n = B_ * 8 * 16 * 9 * 4 * 32;
    k_modw<<<(n + 255) / 256, 256, 0, stream>>>(weight, s, dcoef, wm);
  }
  {
    dim3 grid(HH, 8, B_);
    k_conv<<<grid, 256, 0, stream>>>(wm, xbT, out);
  }
}
